// _NonLocalBlockND_19585050870027
// MI455X (gfx1250) — compile-verified, matrix operand order checked by reading, not run on hardware
//
#include <hip/hip_runtime.h>

// ---------------------------------------------------------------------------
// NonLocalBlockND (softmax-free attention) for MI455X / gfx1250 (wave32, WMMA)
// Key optimization: no softmax => attention is associative:
//   out = (v . k^T / T) . q^T   -- a D x D Gram matrix instead of a T x S
// attention map. Saves ~44 GFLOP and ~170 MB of HBM traffic vs the naive
// ordering. All GEMMs run on v_wmma_f32_16x16x32_bf16 with fp32 accumulate.
// ---------------------------------------------------------------------------

typedef __attribute__((ext_vector_type(16))) __bf16          v16bf;
typedef __attribute__((ext_vector_type(8)))  float           v8f;
typedef unsigned short                                        u16t;
typedef __attribute__((ext_vector_type(8)))  unsigned short  us8;
typedef __attribute__((ext_vector_type(16))) unsigned short  us16;

constexpr int B_ = 16, C_ = 1024, T_ = 2048, D_ = 512, S_ = 1024;
constexpr float EPS_ = 1e-5f;

// fp32 -> bf16 bits, round-to-nearest-even
__device__ __forceinline__ u16t f2bf(float f) {
    union { float f; unsigned u; } cv; cv.f = f;
    unsigned b = cv.u;
    b += 0x7FFFu + ((b >> 16) & 1u);
    return (u16t)(b >> 16);
}

// Load one 16x32 bf16 WMMA fragment ("NT" layout: K contiguous per row).
// CDNA5 16-bit A/B layout: lane L holds row (L&15); VGPRs 0..3 hold
// K = 8*(L>>4)+0..7, VGPRs 4..7 hold K = 16+8*(L>>4)+0..7 -> two contiguous
// 16-byte runs -> two global_load_b128 per lane.
__device__ __forceinline__ v16bf load_frag_nt(const u16t* __restrict__ base,
                                              int ld, int row0, int k0) {
    const int lane = threadIdx.x & 31;
    const int r  = row0 + (lane & 15);
    const int kk = k0 + ((lane >> 4) << 3);
    const us8* p = (const us8*)(base + (size_t)r * ld + kk);
    us8 lo = p[0];
    us8 hi = p[2];   // +16 bf16 elements
    us16 cat = __builtin_shufflevector(lo, hi,
        0,1,2,3,4,5,6,7,8,9,10,11,12,13,14,15);
    return __builtin_bit_cast(v16bf, cat);
}

#define WMMA_BF16(acc, a, b) \
    acc = __builtin_amdgcn_wmma_f32_16x16x32_bf16(false, a, false, b, (short)0, acc, false, false)

// Shared main loop: 8 waves per block arranged 2(M) x 4(N); each wave owns a
// 32x64 tile = 2 A-frags x 4 B-frags -> 8 WMMA per 32-wide K step.
#define GEMM_MAINLOOP(A_, lda_, Bt_, ldb_, m0_, n0_, K_, acc_)               \
    for (int k0 = 0; k0 < (K_); k0 += 32) {                                  \
        v16bf a0 = load_frag_nt(A_,  lda_, (m0_),      k0);                  \
        v16bf a1 = load_frag_nt(A_,  lda_, (m0_) + 16, k0);                  \
        v16bf b0 = load_frag_nt(Bt_, ldb_, (n0_),      k0);                  \
        v16bf b1 = load_frag_nt(Bt_, ldb_, (n0_) + 16, k0);                  \
        v16bf b2 = load_frag_nt(Bt_, ldb_, (n0_) + 32, k0);                  \
        v16bf b3 = load_frag_nt(Bt_, ldb_, (n0_) + 48, k0);                  \
        WMMA_BF16(acc_[0][0], a0, b0);                                       \
        WMMA_BF16(acc_[0][1], a0, b1);                                       \
        WMMA_BF16(acc_[0][2], a0, b2);                                       \
        WMMA_BF16(acc_[0][3], a0, b3);                                       \
        WMMA_BF16(acc_[1][0], a1, b0);                                       \
        WMMA_BF16(acc_[1][1], a1, b1);                                       \
        WMMA_BF16(acc_[1][2], a1, b2);                                       \
        WMMA_BF16(acc_[1][3], a1, b3);                                       \
    }

// ---------------------------------------------------------------------------
// x(B,C,T) fp32 -> xT(B,T,C) bf16, LDS-tiled 32x32 transpose
// ---------------------------------------------------------------------------
__global__ __launch_bounds__(256) void transpose_x_bf16(
        const float* __restrict__ x, u16t* __restrict__ xT) {
    __shared__ float tile[32][33];
    const int b  = blockIdx.z;
    const int t0 = blockIdx.x * 32, c0 = blockIdx.y * 32;
    const int tx = threadIdx.x, ty = threadIdx.y;          // (32, 8)
    const float* xb = x + (size_t)b * C_ * T_;
    #pragma unroll
    for (int j = 0; j < 32; j += 8)
        tile[ty + j][tx] = xb[(size_t)(c0 + ty + j) * T_ + t0 + tx];
    __syncthreads();
    u16t* xTb = xT + (size_t)b * T_ * C_;
    #pragma unroll
    for (int j = 0; j < 32; j += 8)
        xTb[(size_t)(t0 + ty + j) * C_ + c0 + tx] = f2bf(tile[tx][ty + j]);
}

__global__ void convert_bf16(const float* __restrict__ in,
                             u16t* __restrict__ out, int n) {
    int i = blockIdx.x * blockDim.x + threadIdx.x;
    if (i < n) out[i] = f2bf(in[i]);
}

// ---------------------------------------------------------------------------
// Generic NT GEMM: Cm[m,n] = alpha * sum_k A[m,k]*Bt[n,k] (+ biasN[n]), bf16 out
// Grid: (N/256, M/64, batch)
// ---------------------------------------------------------------------------
__global__ __launch_bounds__(256) void gemm_nt_bf16(
        const u16t* __restrict__ A,  long long strideA, int lda,
        const u16t* __restrict__ Bt, long long strideB, int ldb,
        u16t* __restrict__ Cm,       long long strideC, int ldc,
        int K, float alpha, const float* __restrict__ biasN) {
    const int b = blockIdx.z;
    A  += (size_t)strideA * b;
    Bt += (size_t)strideB * b;
    Cm += (size_t)strideC * b;
    const int lane = threadIdx.x & 31;
    const int w = threadIdx.x >> 5;
    const int m0 = blockIdx.y * 64  + (w >> 2) * 32;
    const int n0 = blockIdx.x * 256 + (w & 3) * 64;
    v8f acc[2][4] = {};
    GEMM_MAINLOOP(A, lda, Bt, ldb, m0, n0, K, acc)
    // C/D layout: lane L -> N = L&15; VGPR r -> M = r + 8*(L>>4)
    const int nl = lane & 15, mh = (lane >> 4) << 3;
    #pragma unroll
    for (int i = 0; i < 2; ++i)
        #pragma unroll
        for (int j = 0; j < 4; ++j) {
            const int gn = n0 + j * 16 + nl;
            const float bias = biasN ? biasN[gn] : 0.0f;
            #pragma unroll
            for (int r = 0; r < 8; ++r) {
                const int gm = m0 + i * 16 + mh + r;
                Cm[(size_t)gm * ldc + gn] = f2bf(acc[i][j][r] * alpha + bias);
            }
        }
}

// ---------------------------------------------------------------------------
// K/V projection with fused time-maxpool (kernel=2) via wave32 shfl_xor.
// Out[m=d, n=t] = sum_c W[d,c]*xT[t,c] + biasM[d]; pool adjacent t pairs and
// store (B, D, S) row-major.
// ---------------------------------------------------------------------------
__global__ __launch_bounds__(256) void gemm_nt_pool_bf16(
        const u16t* __restrict__ A,  int lda,
        const u16t* __restrict__ Bt, long long strideB, int ldb,
        u16t* __restrict__ Out,      long long strideOut, int ldOut,
        int K, const float* __restrict__ biasM) {
    const int b = blockIdx.z;
    Bt  += (size_t)strideB  * b;
    Out += (size_t)strideOut * b;
    const int lane = threadIdx.x & 31;
    const int w = threadIdx.x >> 5;
    const int m0 = blockIdx.y * 64  + (w >> 2) * 32;
    const int n0 = blockIdx.x * 256 + (w & 3) * 64;
    v8f acc[2][4] = {};
    GEMM_MAINLOOP(A, lda, Bt, ldb, m0, n0, K, acc)
    const int nl = lane & 15, mh = (lane >> 4) << 3;
    #pragma unroll
    for (int i = 0; i < 2; ++i)
        #pragma unroll
        for (int j = 0; j < 4; ++j) {
            const int gn = n0 + j * 16 + nl;           // t index
            #pragma unroll
            for (int r = 0; r < 8; ++r) {
                const int gm = m0 + i * 16 + mh + r;   // d index
                float val = acc[i][j][r] + biasM[gm];
                // lanes L and L^1 hold adjacent t (N = L&15): pool pairs
                float other = __shfl_xor(val, 1, 32);
                float pooled = fmaxf(val, other);
                if ((gn & 1) == 0)
                    Out[(size_t)gm * ldOut + (gn >> 1)] = f2bf(pooled);
            }
        }
}

// ---------------------------------------------------------------------------
// Output projection: fp32 result + per-row bias (bo[c])
// ---------------------------------------------------------------------------
__global__ __launch_bounds__(256) void gemm_nt_f32(
        const u16t* __restrict__ A,  int lda,
        const u16t* __restrict__ Bt, long long strideB, int ldb,
        float* __restrict__ Y,       long long strideY, int ldc,
        int K, const float* __restrict__ biasM) {
    const int b = blockIdx.z;
    Bt += (size_t)strideB * b;
    Y  += (size_t)strideY * b;
    const int lane = threadIdx.x & 31;
    const int w = threadIdx.x >> 5;
    const int m0 = blockIdx.y * 64  + (w >> 2) * 32;
    const int n0 = blockIdx.x * 256 + (w & 3) * 64;
    v8f acc[2][4] = {};
    GEMM_MAINLOOP(A, lda, Bt, ldb, m0, n0, K, acc)
    const int nl = lane & 15, mh = (lane >> 4) << 3;
    #pragma unroll
    for (int i = 0; i < 2; ++i)
        #pragma unroll
        for (int j = 0; j < 4; ++j) {
            const int gn = n0 + j * 16 + nl;
            #pragma unroll
            for (int r = 0; r < 8; ++r) {
                const int gm = m0 + i * 16 + mh + r;
                Y[(size_t)gm * ldc + gn] = acc[i][j][r] + biasM[gm];
            }
        }
}

// ---------------------------------------------------------------------------
// BatchNorm: per-channel biased mean/var over (B,T); then apply + residual
// ---------------------------------------------------------------------------
__global__ __launch_bounds__(256) void bn_stats(
        const float* __restrict__ y, float* __restrict__ mu,
        float* __restrict__ rsig) {
    const int c = blockIdx.x, tid = threadIdx.x;
    float s1 = 0.f, s2 = 0.f;
    for (int n = tid; n < B_ * T_; n += 256) {
        const int b = n >> 11;          // T_ = 2048
        const int t = n & (T_ - 1);
        const float v = y[((size_t)b * C_ + c) * T_ + t];
        s1 += v; s2 += v * v;
    }
    __shared__ float sh1[256], sh2[256];
    sh1[tid] = s1; sh2[tid] = s2;
    __syncthreads();
    for (int s = 128; s > 0; s >>= 1) {
        if (tid < s) { sh1[tid] += sh1[tid + s]; sh2[tid] += sh2[tid + s]; }
        __syncthreads();
    }
    if (tid == 0) {
        const float inv = 1.0f / (float)(B_ * T_);
        const float m   = sh1[0] * inv;
        const float var = sh2[0] * inv - m * m;
        mu[c] = m;
        rsig[c] = rsqrtf(var + EPS_);
    }
}

__global__ __launch_bounds__(256) void bn_apply_residual(
        const float* __restrict__ y, const float* __restrict__ x,
        const float* __restrict__ gamma, const float* __restrict__ beta,
        const float* __restrict__ mu, const float* __restrict__ rsig,
        float* __restrict__ out) {
    const size_t idx = (size_t)blockIdx.x * blockDim.x + threadIdx.x;
    const int c = (int)((idx / T_) & (C_ - 1));
    out[idx] = gamma[c] * (y[idx] - mu[c]) * rsig[c] + beta[c] + x[idx];
}

// ---------------------------------------------------------------------------
extern "C" void kernel_launch(void* const* d_in, const int* in_sizes, int n_in,
                              void* d_out, int out_size, void* d_ws, size_t ws_size,
                              hipStream_t stream) {
    const float* x     = (const float*)d_in[0];
    const float* Wq    = (const float*)d_in[1];
    const float* bq    = (const float*)d_in[2];
    const float* Wk    = (const float*)d_in[3];
    const float* bk    = (const float*)d_in[4];
    const float* Wv    = (const float*)d_in[5];
    const float* bv    = (const float*)d_in[6];
    const float* Wo    = (const float*)d_in[7];
    const float* bo    = (const float*)d_in[8];
    const float* gamma = (const float*)d_in[9];
    const float* beta  = (const float*)d_in[10];
    float* out = (float*)d_out;

    char* base = (char*)d_ws;
    size_t off = 0;
    auto alloc = [&](size_t bytes) -> void* {
        void* r = base + off;
        off = (off + bytes + 255) & ~(size_t)255;
        return r;
    };
    u16t* xT   = (u16t*)alloc((size_t)B_ * T_ * C_ * 2);   // (B,T,C) bf16
    u16t* Wqb  = (u16t*)alloc((size_t)D_ * C_ * 2);
    u16t* Wkb  = (u16t*)alloc((size_t)D_ * C_ * 2);
    u16t* Wvb  = (u16t*)alloc((size_t)D_ * C_ * 2);
    u16t* Wob  = (u16t*)alloc((size_t)C_ * D_ * 2);
    u16t* qbuf = (u16t*)alloc((size_t)B_ * T_ * D_ * 2);   // (B,T,D)
    u16t* kbuf = (u16t*)alloc((size_t)B_ * D_ * S_ * 2);   // (B,D,S) pooled
    u16t* vbuf = (u16t*)alloc((size_t)B_ * D_ * S_ * 2);   // (B,D,S) pooled
    u16t* Mbuf = (u16t*)alloc((size_t)B_ * D_ * D_ * 2);   // (B,D,D) = v.k^T/T
    u16t* out2 = (u16t*)alloc((size_t)B_ * T_ * D_ * 2);   // (B,T,D)
    float* ybuf = (float*)alloc((size_t)B_ * C_ * T_ * 4); // pre-BN y
    float* mu   = (float*)alloc((size_t)C_ * 4);
    float* rsig = (float*)alloc((size_t)C_ * 4);

    // 1) x -> xT bf16 transpose
    transpose_x_bf16<<<dim3(T_ / 32, C_ / 32, B_), dim3(32, 8), 0, stream>>>(x, xT);

    // 2) weights -> bf16
    const int wn = D_ * C_;
    convert_bf16<<<(wn + 255) / 256, 256, 0, stream>>>(Wq, Wqb, wn);
    convert_bf16<<<(wn + 255) / 256, 256, 0, stream>>>(Wk, Wkb, wn);
    convert_bf16<<<(wn + 255) / 256, 256, 0, stream>>>(Wv, Wvb, wn);
    convert_bf16<<<(wn + 255) / 256, 256, 0, stream>>>(Wo, Wob, wn);

    // 3) q = xT . Wq^T + bq          (M=T, N=D, K=C) -> (B,T,D)
    gemm_nt_bf16<<<dim3(D_ / 256, T_ / 64, B_), 256, 0, stream>>>(
        xT, (long long)T_ * C_, C_, Wqb, 0, C_,
        qbuf, (long long)T_ * D_, D_, C_, 1.0f, bq);

    // 4) k = maxpool2(Wk . x^T + bk) (M=D, N=T, K=C) -> (B,D,S)
    gemm_nt_pool_bf16<<<dim3(T_ / 256, D_ / 64, B_), 256, 0, stream>>>(
        Wkb, C_, xT, (long long)T_ * C_, C_,
        kbuf, (long long)D_ * S_, S_, C_, bk);

    // 5) v = maxpool2(Wv . x^T + bv) (M=D, N=T, K=C) -> (B,D,S)
    gemm_nt_pool_bf16<<<dim3(T_ / 256, D_ / 64, B_), 256, 0, stream>>>(
        Wvb, C_, xT, (long long)T_ * C_, C_,
        vbuf, (long long)D_ * S_, S_, C_, bv);

    // 6) M = (v . k^T) / T           (M=D, N=D, K=S) -> (B,D,D)
    //    (reassociated attention: out = M . q^T, valid since no softmax)
    gemm_nt_bf16<<<dim3(D_ / 256, D_ / 64, B_), 256, 0, stream>>>(
        vbuf, (long long)D_ * S_, S_, kbuf, (long long)D_ * S_, S_,
        Mbuf, (long long)D_ * D_, D_, S_, 1.0f / (float)T_, nullptr);

    // 7) out2[t,d] = sum_d' q[t,d'] * M[d,d']  (M=T, N=D, K=D) -> (B,T,D)
    gemm_nt_bf16<<<dim3(D_ / 256, T_ / 64, B_), 256, 0, stream>>>(
        qbuf, (long long)T_ * D_, D_, Mbuf, (long long)D_ * D_, D_,
        out2, (long long)T_ * D_, D_, D_, 1.0f, nullptr);

    // 8) y = Wo . out2^T + bo        (M=C, N=T, K=D) -> fp32 (B,C,T)
    gemm_nt_f32<<<dim3(T_ / 256, C_ / 64, B_), 256, 0, stream>>>(
        Wob, D_, out2, (long long)T_ * D_, D_,
        ybuf, (long long)C_ * T_, T_, D_, bo);

    // 9) BN stats + apply + residual
    bn_stats<<<C_, 256, 0, stream>>>(ybuf, mu, rsig);
    const size_t total = (size_t)B_ * C_ * T_;
    bn_apply_residual<<<(unsigned)(total / 256), 256, 0, stream>>>(
        ybuf, x, gamma, beta, mu, rsig, out);
}
